// LayerNormLSTMCell_8873402433922
// MI455X (gfx1250) — compile-verified, matrix operand order checked by reading, not run on hardware
//
#include <hip/hip_runtime.h>
#include <hip/hip_bf16.h>
#include <stdint.h>

typedef __attribute__((ext_vector_type(16))) __bf16 v16bf;
typedef __attribute__((ext_vector_type(8)))  float  v8f;

#define GM 4096
#define GN 4096
#define GK 1024
#define TK 32
#define LSTRIDE 56                    // bf16 elems per LDS row: 112B, 16B-aligned, bank-conflict-free
#define LBYTES  (128 * LSTRIDE * 2)   // bytes per tile buffer (14336)

// ---------------------------------------------------------------------------
// CDNA5 async global->LDS copy (ASYNCcnt path), ISA 15.18.3 opcode 98.
// vdst = per-lane LDS byte address, vaddr = per-lane 64-bit global address (GV).
// ---------------------------------------------------------------------------
__device__ __forceinline__ void async_copy16(uint32_t lds, uint64_t gaddr) {
    asm volatile("global_load_async_to_lds_b128 %0, %1, off"
                 :: "v"(lds), "v"(gaddr) : "memory");
}
__device__ __forceinline__ void wait_async_le4() {
    asm volatile("s_wait_asynccnt 0x4" ::: "memory");
}
__device__ __forceinline__ void wait_async_0() {
    asm volatile("s_wait_asynccnt 0x0" ::: "memory");
}

// ---------------------------------------------------------------------------
// pack f32 -> bf16 (row-major, elementwise)
// ---------------------------------------------------------------------------
__global__ __launch_bounds__(256)
void pack_bf16(const float* __restrict__ in, __bf16* __restrict__ out) {
    size_t i = (size_t)blockIdx.x * 256 + threadIdx.x;
    float4 v = *(const float4*)&in[i * 4];
    union { __bf16 b[4]; uint2 u; } o;
    o.b[0] = (__bf16)v.x; o.b[1] = (__bf16)v.y;
    o.b[2] = (__bf16)v.z; o.b[3] = (__bf16)v.w;
    *(uint2*)&out[i * 4] = o.u;
}

// ---------------------------------------------------------------------------
// W[k][n] f32  ->  Wt[n][k] bf16  (tiled 32x32 transpose via LDS)
// ---------------------------------------------------------------------------
__global__ __launch_bounds__(256)
void transpose_pack_bf16(const float* __restrict__ in, __bf16* __restrict__ out) {
    __shared__ __bf16 t[32][33];
    const int n0 = blockIdx.x * 32, k0 = blockIdx.y * 32;
    const int x = threadIdx.x & 31, y0 = threadIdx.x >> 5;   // y0: 0..7
#pragma unroll
    for (int i = 0; i < 4; ++i) {
        int y = y0 + i * 8;
        t[y][x] = (__bf16)in[(size_t)(k0 + y) * GN + n0 + x];
    }
    __syncthreads();
#pragma unroll
    for (int i = 0; i < 4; ++i) {
        int y = y0 + i * 8;                       // local n
        out[(size_t)(n0 + y) * GK + k0 + x] = t[x][y];
    }
}

// ---------------------------------------------------------------------------
// GEMM: Z[4096,4096] = Abf[4096,1024] * Wt[4096,1024]^T + bias
// bf16 WMMA, f32 accum, async double-buffered LDS staging.
// ---------------------------------------------------------------------------
__global__ __launch_bounds__(256)
void gemm_bias_bf16wmma(const __bf16* __restrict__ Abf,   // [M][K] row-major
                        const __bf16* __restrict__ Wt,    // [N][K] row-major (W transposed)
                        const float*  __restrict__ bias,  // [N]
                        float*        __restrict__ Z)     // [M][N]
{
    __shared__ __attribute__((aligned(16))) __bf16 Alds[2][128 * LSTRIDE];
    __shared__ __attribute__((aligned(16))) __bf16 Blds[2][128 * LSTRIDE];

    const int tid  = threadIdx.x;
    const int lane = tid & 31, wave = tid >> 5;
    const int lm = lane & 15,  lh = lane >> 4;
    const int wave_m = wave >> 2;   // 0..1 -> 64 rows
    const int wave_n = wave & 3;    // 0..3 -> 32 cols
    const int block_m = blockIdx.y * 128;
    const int block_n = blockIdx.x * 128;

    // ---- per-thread async copy chunks: 2x16B for A, 2x16B for B per K-tile
    uint64_t gaA[2], gaB[2];
    uint32_t ldA[2], ldB[2];
    const uint32_t aBase = (uint32_t)(uintptr_t)(&Alds[0][0]);
    const uint32_t bBase = (uint32_t)(uintptr_t)(&Blds[0][0]);
#pragma unroll
    for (int i = 0; i < 2; ++i) {
        int c = tid + i * 256;        // 0..511 chunks of 16B (128 rows x 4 quads)
        int row = c >> 2, q = c & 3;
        gaA[i] = (uint64_t)(uintptr_t)(Abf + (((size_t)(block_m + row)) << 10) + q * 8);
        gaB[i] = (uint64_t)(uintptr_t)(Wt  + (((size_t)(block_n + row)) << 10) + q * 8);
        ldA[i] = aBase + row * (LSTRIDE * 2) + q * 16;
        ldB[i] = bBase + row * (LSTRIDE * 2) + q * 16;
    }

    auto copy_tile = [&](int buf) {
#pragma unroll
        for (int i = 0; i < 2; ++i) {
            async_copy16(ldA[i] + buf * LBYTES, gaA[i]);
            async_copy16(ldB[i] + buf * LBYTES, gaB[i]);
            gaA[i] += TK * 2;         // advance 32 bf16 in K
            gaB[i] += TK * 2;
        }
    };

    v8f acc[4][2];
    const v8f vz = {};
#pragma unroll
    for (int r = 0; r < 4; ++r)
#pragma unroll
        for (int c = 0; c < 2; ++c) acc[r][c] = vz;

    copy_tile(0);                     // prologue: 4 async in flight

    for (int kt = 0; kt < GK / TK; ++kt) {
        const int cur = kt & 1;
        if (kt + 1 < GK / TK) { copy_tile(cur ^ 1); wait_async_le4(); }
        else                  { wait_async_0(); }
        __syncthreads();

        // fragments (ISA 16-bit layout: chunk0 K=lh*8.., chunk1 K=16+lh*8..)
        v16bf afr[4], bfr[2];
#pragma unroll
        for (int r = 0; r < 4; ++r) {
            int m = wave_m * 64 + r * 16 + lm;
            union { v16bf v; uint4 q[2]; } u;
            u.q[0] = *(const uint4*)&Alds[cur][m * LSTRIDE + lh * 8];
            u.q[1] = *(const uint4*)&Alds[cur][m * LSTRIDE + 16 + lh * 8];
            afr[r] = u.v;
        }
#pragma unroll
        for (int c = 0; c < 2; ++c) {
            int n = wave_n * 32 + c * 16 + lm;
            union { v16bf v; uint4 q[2]; } u;
            u.q[0] = *(const uint4*)&Blds[cur][n * LSTRIDE + lh * 8];
            u.q[1] = *(const uint4*)&Blds[cur][n * LSTRIDE + 16 + lh * 8];
            bfr[c] = u.v;
        }

#pragma unroll
        for (int r = 0; r < 4; ++r)
#pragma unroll
            for (int c = 0; c < 2; ++c)
                acc[r][c] = __builtin_amdgcn_wmma_f32_16x16x32_bf16(
                    false, afr[r], false, bfr[c], (short)0, acc[r][c], false, false);
        __syncthreads();
    }

    // ---- epilogue: D layout -> VGPR e holds m = e + 8*lh, n = lm.
    // N hardcoded -> stores use immediate offsets e*16384B.
#pragma unroll
    for (int c = 0; c < 2; ++c) {
        int col = block_n + wave_n * 32 + c * 16 + lm;
        float bb = bias[col];
#pragma unroll
        for (int r = 0; r < 4; ++r) {
            int row0 = block_m + wave_m * 64 + r * 16 + 8 * lh;
            float* p = Z + (((size_t)row0) << 12) + col;
#pragma unroll
            for (int e = 0; e < 8; ++e)
                p[(size_t)e << 12] = acc[r][c][e] + bb;
        }
    }
}

// ---------------------------------------------------------------------------
// Fused LN(Z1)+LN(Z2) -> gates -> cell LN -> h,c
// ---------------------------------------------------------------------------
__device__ inline float sigmoidf_(float x) { return 1.0f / (1.0f + __expf(-x)); }

__device__ inline void block_reduce4(float& a, float& b, float& c, float& d,
                                     float* red, int tid)
{
    __syncthreads();
#pragma unroll
    for (int off = 16; off > 0; off >>= 1) {
        a += __shfl_down(a, off, 32);
        b += __shfl_down(b, off, 32);
        c += __shfl_down(c, off, 32);
        d += __shfl_down(d, off, 32);
    }
    int wid = tid >> 5;
    if ((tid & 31) == 0) {
        red[wid * 4 + 0] = a; red[wid * 4 + 1] = b;
        red[wid * 4 + 2] = c; red[wid * 4 + 3] = d;
    }
    __syncthreads();
    if (tid == 0) {
        float ra = 0, rb = 0, rc = 0, rd = 0;
#pragma unroll
        for (int w = 0; w < 8; ++w) {
            ra += red[w * 4 + 0]; rb += red[w * 4 + 1];
            rc += red[w * 4 + 2]; rd += red[w * 4 + 3];
        }
        red[32] = ra; red[33] = rb; red[34] = rc; red[35] = rd;
    }
    __syncthreads();
    a = red[32]; b = red[33]; c = red[34]; d = red[35];
}

__global__ __launch_bounds__(256)
void lstm_fuse(const float* __restrict__ Z1, const float* __restrict__ Z2,
               const float* __restrict__ c_prev,
               const float* __restrict__ g_i2h, const float* __restrict__ be_i2h,
               const float* __restrict__ g_h2h, const float* __restrict__ be_h2h,
               const float* __restrict__ g_c,   const float* __restrict__ be_c,
               float* __restrict__ h_out, float* __restrict__ c_out)
{
    constexpr int NN = 4096;
    constexpr int H  = 1024;
    constexpr float EPS = 1e-6f;

    __shared__ float z1row[NN];
    __shared__ float z2row[NN];
    __shared__ float red[36];

    const int b   = blockIdx.x;
    const int tid = threadIdx.x;

    float s1 = 0, q1 = 0, s2 = 0, q2 = 0;
#pragma unroll
    for (int i = 0; i < 4; ++i) {
        int idx = (tid + i * 256) * 4;
        float4 v1 = *(const float4*)&Z1[(size_t)b * NN + idx];
        float4 v2 = *(const float4*)&Z2[(size_t)b * NN + idx];
        *(float4*)&z1row[idx] = v1;
        *(float4*)&z2row[idx] = v2;
        s1 += v1.x + v1.y + v1.z + v1.w;
        q1 += v1.x * v1.x + v1.y * v1.y + v1.z * v1.z + v1.w * v1.w;
        s2 += v2.x + v2.y + v2.z + v2.w;
        q2 += v2.x * v2.x + v2.y * v2.y + v2.z * v2.z + v2.w * v2.w;
    }
    block_reduce4(s1, q1, s2, q2, red, tid);

    const float mean1 = s1 * (1.0f / NN);
    const float var1  = fmaxf((q1 - NN * mean1 * mean1) * (1.0f / (NN - 1)), 0.0f);
    const float inv1  = 1.0f / (sqrtf(var1) + EPS);
    const float mean2 = s2 * (1.0f / NN);
    const float var2  = fmaxf((q2 - NN * mean2 * mean2) * (1.0f / (NN - 1)), 0.0f);
    const float inv2  = 1.0f / (sqrtf(var2) + EPS);

    float crv[4], aov[4];
    float sc = 0, qc = 0;
#pragma unroll
    for (int t = 0; t < 4; ++t) {
        int j = tid + t * 256;
        int ni = j, nf = j + H, nu = j + 2 * H, no = j + 3 * H;
        float hi_i = g_i2h[ni] * (z1row[ni] - mean1) * inv1 + be_i2h[ni];
        float hi_f = g_i2h[nf] * (z1row[nf] - mean1) * inv1 + be_i2h[nf];
        float hi_u = g_i2h[nu] * (z1row[nu] - mean1) * inv1 + be_i2h[nu];
        float hi_o = g_i2h[no] * (z1row[no] - mean1) * inv1 + be_i2h[no];
        float hh_i = g_h2h[ni] * (z2row[ni] - mean2) * inv2 + be_h2h[ni];
        float hh_f = g_h2h[nf] * (z2row[nf] - mean2) * inv2 + be_h2h[nf];
        float hh_u = g_h2h[nu] * (z2row[nu] - mean2) * inv2 + be_h2h[nu];
        float hh_o = g_h2h[no] * (z2row[no] - mean2) * inv2 + be_h2h[no];
        float ai = hi_i + hh_i, af = hi_f + hh_f;
        float au = hi_u + hh_u, ao = hi_o + hh_o;
        float cr = c_prev[(size_t)b * H + j] * sigmoidf_(af + 1.0f)
                 + tanhf(au) * sigmoidf_(ai);
        crv[t] = cr; aov[t] = ao;
        sc += cr; qc += cr * cr;
    }

    float d0 = 0, d1 = 0;
    block_reduce4(sc, qc, d0, d1, red, tid);

    const float meanc = sc * (1.0f / H);
    const float varc  = fmaxf((qc - H * meanc * meanc) * (1.0f / (H - 1)), 0.0f);
    const float invc  = 1.0f / (sqrtf(varc) + EPS);

#pragma unroll
    for (int t = 0; t < 4; ++t) {
        int j = tid + t * 256;
        float cn = g_c[j] * (crv[t] - meanc) * invc + be_c[j];
        c_out[(size_t)b * H + j] = cn;
        h_out[(size_t)b * H + j] = sigmoidf_(aov[t]) * tanhf(cn);
    }
}

// ---------------------------------------------------------------------------
extern "C" void kernel_launch(void* const* d_in, const int* in_sizes, int n_in,
                              void* d_out, int out_size, void* d_ws, size_t ws_size,
                              hipStream_t stream)
{
    (void)in_sizes; (void)n_in; (void)out_size; (void)ws_size;
    const float* input  = (const float*)d_in[0];
    const float* h_prev = (const float*)d_in[1];
    const float* c_prev = (const float*)d_in[2];
    const float* w_i2h  = (const float*)d_in[3];
    const float* b_i2h  = (const float*)d_in[4];
    const float* w_h2h  = (const float*)d_in[5];
    const float* b_h2h  = (const float*)d_in[6];
    const float* g_i2h  = (const float*)d_in[7];
    const float* be_i2h = (const float*)d_in[8];
    const float* g_h2h  = (const float*)d_in[9];
    const float* be_h2h = (const float*)d_in[10];
    const float* g_c    = (const float*)d_in[11];
    const float* be_c   = (const float*)d_in[12];

    char* ws = (char*)d_ws;
    float*  Z1  = (float*) (ws);                          // 64 MiB
    float*  Z2  = (float*) (ws + ((size_t)64  << 20));    // 64 MiB
    __bf16* Abf = (__bf16*)(ws + ((size_t)128 << 20));    // 8 MiB
    __bf16* Hbf = (__bf16*)(ws + ((size_t)136 << 20));    // 8 MiB
    __bf16* Wt1 = (__bf16*)(ws + ((size_t)144 << 20));    // 8 MiB
    __bf16* Wt2 = (__bf16*)(ws + ((size_t)152 << 20));    // 8 MiB

    // precision conversion passes (bf16 + W transpose)
    pack_bf16<<<4096, 256, 0, stream>>>(input,  Abf);
    pack_bf16<<<4096, 256, 0, stream>>>(h_prev, Hbf);
    transpose_pack_bf16<<<dim3(GN / 32, GK / 32), 256, 0, stream>>>(w_i2h, Wt1);
    transpose_pack_bf16<<<dim3(GN / 32, GK / 32), 256, 0, stream>>>(w_h2h, Wt2);

    dim3 grid(GN / 128, GM / 128), blk(256);
    gemm_bias_bf16wmma<<<grid, blk, 0, stream>>>(Abf, Wt1, b_i2h, Z1);
    gemm_bias_bf16wmma<<<grid, blk, 0, stream>>>(Hbf, Wt2, b_h2h, Z2);

    float* h_out = (float*)d_out;
    float* c_out = h_out + (size_t)GM * 1024;
    lstm_fuse<<<GM, 256, 0, stream>>>(Z1, Z2, c_prev, g_i2h, be_i2h,
                                      g_h2h, be_h2h, g_c, be_c, h_out, c_out);
}
